// HGAN_1580547975054
// MI455X (gfx1250) — matrix operand order repaired and checked by reading, not compiled, not run on hardware
//
#include <hip/hip_runtime.h>
#include <hip/hip_bf16.h>

// ---------------------------------------------------------------------------
// HGAN triplet loss on gfx1250 (MI455X).
// B=32, N=M=256, E=300 (padded to 320), H=256, K=8.
// All GEMMs on v_wmma_f32_16x16x32_f16.  Per-(b,k) att map in LDS (f16).
// Attention operands register/LDS-cached so the kernel is WMMA-bound.
// ---------------------------------------------------------------------------

#define B_   32
#define N_   256
#define E_   300
#define EP_  320          // E padded to multiple of 32 for WMMA K
#define H_   256
#define K_   8
#define MARGIN_ 0.2f

typedef _Float16 v16h __attribute__((ext_vector_type(16)));
typedef _Float16 v8h  __attribute__((ext_vector_type(8)));
typedef float    v8f  __attribute__((ext_vector_type(8)));

#define ATT_PITCH 264     // halves; 528B rows -> 16B aligned, conflict-friendly
#define BST_PITCH 264

// ---------------------------------------------------------------------------
// f32 [rows,300] -> f16 [rows,320], zero-padded tail. One WG per row.
// ---------------------------------------------------------------------------
__global__ __launch_bounds__(256) void convert_pad_kernel(
    const float* __restrict__ src, _Float16* __restrict__ dst)
{
    const int row = blockIdx.x;
    for (int e = threadIdx.x; e < EP_; e += 256)
        dst[(size_t)row * EP_ + e] =
            (e < E_) ? (_Float16)src[(size_t)row * E_ + e] : (_Float16)0.0f;
}

// ---------------------------------------------------------------------------
// Projection GEMM: P[r,h] = sum_e xh[r,e]*Wh[h,e] + bias[h]   (WMMA, K=e)
//   r = b*N+n.  Writes P [B,N,H] f16 and (optionally) PT [B,H,N] f16.
// One 16x16 tile per wave, 8 waves/WG.  10 wmma per tile (EP=320).
// ---------------------------------------------------------------------------
__global__ __launch_bounds__(256) void proj_wmma_kernel(
    const _Float16* __restrict__ xh,   // [B*N, EP]
    const _Float16* __restrict__ Wh,   // [H,   EP]
    const float*    __restrict__ bias, // [H]
    _Float16* __restrict__ P,          // [B,N,H]
    _Float16* __restrict__ PT)         // [B,H,N] or nullptr
{
    __shared__ _Float16 st[8][16][16];
    const int tid  = threadIdx.x;
    const int wave = tid >> 5;
    const int lane = tid & 31;
    const int ln   = lane & 15;
    const int hi   = lane >> 4;

    const int t  = blockIdx.x * 8 + wave;   // tile id over (rows/16) x (H/16)
    const int n0 = (t >> 4) << 4;           // global row tile (b*N+n space)
    const int h0 = (t & 15) << 4;           // h tile

    v8f c = {};
    const _Float16* xrow = xh + (size_t)(n0 + ln) * EP_;
    const _Float16* wrow = Wh + (size_t)(h0 + ln) * EP_;
#pragma unroll
    for (int e0 = 0; e0 < EP_; e0 += 32) {
        const int eA = e0 + hi * 16;
        v8h a0 = *(const v8h*)(xrow + eA);
        v8h a1 = *(const v8h*)(xrow + eA + 8);
        v16h a;
#pragma unroll
        for (int i = 0; i < 8; ++i) { a[i] = a0[i]; a[i + 8] = a1[i]; }
        v16h bb = *(const v16h*)(wrow + e0 + hi * 16);
        c = __builtin_amdgcn_wmma_f32_16x16x32_f16(
                false, a, false, bb, (short)0, c, false, false);
    }

    const float bv = bias[h0 + ln];
#pragma unroll
    for (int i = 0; i < 8; ++i)
        st[wave][i + hi * 8][ln] = (_Float16)(c[i] + bv);

    // coalesced writeback: 32 lanes cover 16 rows x 2 half-rows of 8 halves
    const int r  = lane >> 1;
    const int cc = (lane & 1) * 8;
    v8h rowv = *(const v8h*)&st[wave][r][cc];
    *(v8h*)&P[(size_t)(n0 + r) * H_ + h0 + cc] = rowv;

    if (PT) {
        const int b  = n0 >> 8;          // 16-row tile lies within one b
        const int nn = n0 & 255;
        v8h tv;
#pragma unroll
        for (int j = 0; j < 8; ++j) tv[j] = st[wave][cc + j][r];
        *(v8h*)&PT[((size_t)(b * H_ + h0 + r)) * N_ + nn + cc] = tv;
    }
}

// ---------------------------------------------------------------------------
// Fused attention per (b,k,side):
//   logits[n,m] = sum_h (Pa[n,h]*Watt[k,h]) * Pc[m,h] + batt[k]   (WMMA)
//   att = softmax over flattened (n,m)                            (LDS)
//   R[n,h] = sum_m att[n,m]*PcT[h,m]                              (WMMA)
//   out[b,k,h] = (1/Z) * sum_n Pa[n,h]*R[n,h]                     (PaT fold)
// grid = (B*K, 2) WGs of 256 threads (8 waves); y = pos/neg side.
// Pass1: A (wk-scaled) register-resident per wave, B panel staged in LDS.
// Pass3: B register-resident per h-tile, A streamed from LDS att map.
// ---------------------------------------------------------------------------
#define ATTN_SMEM_BYTES (2560 + 16 * BST_PITCH * 2 + (size_t)N_ * ATT_PITCH * 2)

__global__ __launch_bounds__(256) void attn_kernel(
    const _Float16* __restrict__ Pa,    // [B,N,H]
    const _Float16* __restrict__ PaT,   // [B,H,N]
    const _Float16* __restrict__ Pp,    // [B,M,H]
    const _Float16* __restrict__ PpT,   // [B,H,M]
    const _Float16* __restrict__ Pn,    // [B,M,H]
    const _Float16* __restrict__ PnT,   // [B,H,M]
    const float*    __restrict__ Watt,  // [K,H]
    const float*    __restrict__ batt,  // [K]
    float*          __restrict__ outp,  // [B,K,H]
    float*          __restrict__ outn)  // [B,K,H]
{
    extern __shared__ char smem[];
    float*    outacc = (float*)smem;                   // 256 f32
    float*    red    = (float*)(smem + 1024);          // 256 f32
    _Float16* wk     = (_Float16*)(smem + 2048);       // 256 f16
    _Float16* bst    = (_Float16*)(smem + 2560);       // 16 x BST_PITCH f16
    _Float16* attL   = (_Float16*)(smem + 2560 + 16 * BST_PITCH * 2);

    const int side = blockIdx.y;
    const _Float16* Pc   = side ? Pn  : Pp;
    const _Float16* PcT  = side ? PnT : PpT;
    float*          outg = side ? outn : outp;

    const int bk   = blockIdx.x;
    const int b    = bk / K_;
    const int k    = bk % K_;
    const int tid  = threadIdx.x;
    const int wave = tid >> 5;
    const int lane = tid & 31;
    const int ln   = lane & 15;        // tile row (A) / tile col (B,C)
    const int hi   = lane >> 4;        // lane-half selector

    wk[tid]     = (_Float16)Watt[k * H_ + tid];
    outacc[tid] = 0.0f;
    __syncthreads();

    // ---- Pass 1: logits GEMM + block max --------------------------------
    // Cache wk-scaled A fragments for this wave's two n-rows (tiles wave, wave+8)
    v16h aReg[2][8];
#pragma unroll
    for (int rr = 0; rr < 2; ++rr) {
        const int n0 = (wave + rr * 8) << 4;
        const _Float16* parow = Pa + ((size_t)(b * N_ + n0 + ln)) * H_;
#pragma unroll
        for (int kk = 0; kk < 8; ++kk) {
            const int hA = kk * 32 + hi * 16;
            v8h a0 = *(const v8h*)(parow + hA);
            v8h a1 = *(const v8h*)(parow + hA + 8);
            v16h a;
#pragma unroll
            for (int i = 0; i < 8; ++i) {
                a[i]     = a0[i] * wk[hA + i];
                a[i + 8] = a1[i] * wk[hA + 8 + i];
            }
            aReg[rr][kk] = a;
        }
    }

    const float batt_k = batt[k];
    float lmax = -3.0e38f;
    const int srow = tid >> 4;           // staging: row 0..15
    const int scol = (tid & 15) * 16;    // staging: 16-half chunk
#pragma unroll 1
    for (int mt = 0; mt < 16; ++mt) {
        const int m0 = mt << 4;
        __syncthreads();   // previous iteration's bst readers done
        *(v16h*)&bst[srow * BST_PITCH + scol] =
            *(const v16h*)&Pc[((size_t)(b * N_ + m0 + srow)) * H_ + scol];
        __syncthreads();
#pragma unroll
        for (int rr = 0; rr < 2; ++rr) {
            const int n0 = (wave + rr * 8) << 4;
            v8f c = {};
#pragma unroll
            for (int kk = 0; kk < 8; ++kk) {
                v16h bb = *(const v16h*)&bst[ln * BST_PITCH + kk * 32 + hi * 16];
                c = __builtin_amdgcn_wmma_f32_16x16x32_f16(
                        false, aReg[rr][kk], false, bb, (short)0, c, false, false);
            }
#pragma unroll
            for (int i = 0; i < 8; ++i) {
                const float v = c[i] + batt_k;
                const int n = n0 + i + hi * 8;
                attL[n * ATT_PITCH + m0 + ln] = (_Float16)v;
                lmax = fmaxf(lmax, v);
            }
        }
    }
    red[tid] = lmax;
    __syncthreads();
#pragma unroll
    for (int s = 128; s > 0; s >>= 1) {
        if (tid < s) red[tid] = fmaxf(red[tid], red[tid + s]);
        __syncthreads();
    }
    const float gmax = red[0];
    __syncthreads();

    // ---- Pass 2: exp + sum ----------------------------------------------
    float lsum = 0.0f;
    for (int idx = tid; idx < N_ * N_; idx += 256) {
        const int n = idx >> 8, m = idx & 255;
        const float e = __expf((float)attL[n * ATT_PITCH + m] - gmax);
        attL[n * ATT_PITCH + m] = (_Float16)e;
        lsum += e;
    }
    red[tid] = lsum;
    __syncthreads();
#pragma unroll
    for (int s = 128; s > 0; s >>= 1) {
        if (tid < s) red[tid] += red[tid + s];
        __syncthreads();
    }
    const float Z = red[0];
    __syncthreads();

    // ---- Pass 3: R = att @ PcT, fold with PaT ----------------------------
#pragma unroll 1
    for (int ht = wave; ht < 16; ht += 8) {
        const int h0 = ht << 4;
        const _Float16* pct  = PcT + ((size_t)(b * H_ + h0 + ln)) * N_;
        const _Float16* patr = PaT + ((size_t)(b * H_ + h0 + ln)) * N_;
        v16h bReg[8];
#pragma unroll
        for (int kk = 0; kk < 8; ++kk)
            bReg[kk] = *(const v16h*)(pct + kk * 32 + hi * 16);

        float facc = 0.0f;
#pragma unroll 1
        for (int nt = 0; nt < 16; ++nt) {
            const int n0 = nt << 4;
            const _Float16* arow = attL + (n0 + ln) * ATT_PITCH;
            v8f c = {};
#pragma unroll
            for (int kk = 0; kk < 8; ++kk) {
                const int mA = kk * 32 + hi * 16;
                v8h a0 = *(const v8h*)(arow + mA);
                v8h a1 = *(const v8h*)(arow + mA + 8);
                v16h a;
#pragma unroll
                for (int i = 0; i < 8; ++i) { a[i] = a0[i]; a[i + 8] = a1[i]; }
                c = __builtin_amdgcn_wmma_f32_16x16x32_f16(
                        false, a, false, bReg[kk], (short)0, c, false, false);
            }
            v8h pav = *(const v8h*)(patr + n0 + hi * 8);
#pragma unroll
            for (int i = 0; i < 8; ++i)
                facc += c[i] * (float)pav[i];
        }
        atomicAdd(&outacc[h0 + ln], facc);
    }
    __syncthreads();
    outg[(size_t)bk * H_ + tid] = outacc[tid] / Z;
}

// ---------------------------------------------------------------------------
// score[side,b] = sum_{k,h} out[side][b,k,h] * W_fc[k*H+h] + b_fc
// ---------------------------------------------------------------------------
__global__ __launch_bounds__(256) void score_kernel(
    const float* __restrict__ out_pos, const float* __restrict__ out_neg,
    const float* __restrict__ Wfc, const float* __restrict__ bfc,
    float* __restrict__ scores)
{
    __shared__ float red[256];
    const int side = blockIdx.x >> 5;
    const int b    = blockIdx.x & 31;
    const int tid  = threadIdx.x;
    const float* src = (side == 0 ? out_pos : out_neg) + (size_t)b * K_ * H_;

    float s = 0.0f;
#pragma unroll
    for (int idx = tid; idx < K_ * H_; idx += 256)
        s = fmaf(src[idx], Wfc[idx], s);
    red[tid] = s;
    __syncthreads();
#pragma unroll
    for (int st = 128; st > 0; st >>= 1) {
        if (tid < st) red[tid] += red[tid + st];
        __syncthreads();
    }
    if (tid == 0) scores[side * B_ + b] = red[0] + bfc[0];
}

__global__ void loss_kernel(const float* __restrict__ scores,
                            float* __restrict__ out)
{
    __shared__ float red[32];
    const int tid = threadIdx.x;
    const float v = fmaxf(0.0f, scores[B_ + tid] - scores[tid] + MARGIN_);
    red[tid] = v;
    __syncthreads();
    if (tid == 0) {
        float s = 0.0f;
        for (int i = 0; i < B_; ++i) s += red[i];
        out[0] = s / (float)B_;
    }
}

// ---------------------------------------------------------------------------
extern "C" void kernel_launch(void* const* d_in, const int* in_sizes, int n_in,
                              void* d_out, int out_size, void* d_ws, size_t ws_size,
                              hipStream_t stream)
{
    (void)in_sizes; (void)n_in; (void)out_size; (void)ws_size;

    const float* he_anchor = (const float*)d_in[0];
    const float* he_pos    = (const float*)d_in[1];
    const float* he_neg    = (const float*)d_in[2];
    const float* W_a2h     = (const float*)d_in[3];
    const float* b_a2h     = (const float*)d_in[4];
    const float* W_c2h     = (const float*)d_in[5];
    const float* b_c2h     = (const float*)d_in[6];
    const float* W_att     = (const float*)d_in[7];
    const float* b_att     = (const float*)d_in[8];
    const float* W_fc      = (const float*)d_in[9];
    const float* b_fc      = (const float*)d_in[10];
    float* out = (float*)d_out;

    // workspace carve-up (all chunks 16B aligned)
    const size_t ROWS = (size_t)B_ * N_;             // 8192
    const size_t PN   = ROWS * H_;                   // 2,097,152 elems
    char* ws = (char*)d_ws;
    _Float16* xh_a = (_Float16*)ws;  ws += ROWS * EP_ * 2;   // 5.24 MB
    _Float16* xh_p = (_Float16*)ws;  ws += ROWS * EP_ * 2;
    _Float16* xh_n = (_Float16*)ws;  ws += ROWS * EP_ * 2;
    _Float16* Wh_a = (_Float16*)ws;  ws += (size_t)H_ * EP_ * 2;  // 160 KB
    _Float16* Wh_c = (_Float16*)ws;  ws += (size_t)H_ * EP_ * 2;
    _Float16* P_a  = (_Float16*)ws;  ws += PN * 2;   // 4 MB each
    _Float16* P_aT = (_Float16*)ws;  ws += PN * 2;
    _Float16* P_p  = (_Float16*)ws;  ws += PN * 2;
    _Float16* P_pT = (_Float16*)ws;  ws += PN * 2;
    _Float16* P_n  = (_Float16*)ws;  ws += PN * 2;
    _Float16* P_nT = (_Float16*)ws;  ws += PN * 2;
    float* out_pos = (float*)ws;     ws += (size_t)B_ * K_ * H_ * 4;
    float* out_neg = (float*)ws;     ws += (size_t)B_ * K_ * H_ * 4;
    float* scores  = (float*)ws;     ws += 64 * 4;

    (void)hipFuncSetAttribute((const void*)attn_kernel,
                              hipFuncAttributeMaxDynamicSharedMemorySize,
                              (int)ATTN_SMEM_BYTES);

    // f32 -> f16 (pad E 300->320)
    convert_pad_kernel<<<(int)ROWS, 256, 0, stream>>>(he_anchor, xh_a);
    convert_pad_kernel<<<(int)ROWS, 256, 0, stream>>>(he_pos,    xh_p);
    convert_pad_kernel<<<(int)ROWS, 256, 0, stream>>>(he_neg,    xh_n);
    convert_pad_kernel<<<H_, 256, 0, stream>>>(W_a2h, Wh_a);
    convert_pad_kernel<<<H_, 256, 0, stream>>>(W_c2h, Wh_c);

    // projections via WMMA: 8192 tiles, 8 waves/WG -> 1024 WGs
    const int projGrid = (int)(ROWS / 16) * (H_ / 16) / 8;
    proj_wmma_kernel<<<projGrid, 256, 0, stream>>>(xh_a, Wh_a, b_a2h, P_a, P_aT);
    proj_wmma_kernel<<<projGrid, 256, 0, stream>>>(xh_p, Wh_c, b_c2h, P_p, P_pT);
    proj_wmma_kernel<<<projGrid, 256, 0, stream>>>(xh_n, Wh_c, b_c2h, P_n, P_nT);

    // fused attention, one WG per (b,k), both sides in one launch
    attn_kernel<<<dim3(B_ * K_, 2), 256, ATTN_SMEM_BYTES, stream>>>(
        P_a, P_aT, P_p, P_pT, P_n, P_nT, W_att, b_att, out_pos, out_neg);

    // scores + margin loss
    score_kernel<<<64, 256, 0, stream>>>(out_pos, out_neg, W_fc, b_fc, scores);
    loss_kernel<<<1, 32, 0, stream>>>(scores, out);
}
